// VMambaFusionModule_58643483459627
// MI455X (gfx1250) — compile-verified, matrix operand order checked by reading, not run on hardware
//
#include <hip/hip_runtime.h>
#include <hip/hip_bf16.h>
#include <math.h>

// ---------------- problem constants ----------------
#define Bc     8
#define Lc     4096
#define HIDc   1024
#define TDIMc  768
#define LTc    77
#define DIMc   512
#define DIN    1024      // D_INNER
#define NH     32        // NHEADS
#define HD     32        // HEADDIM
#define DSTATE 64
#define CK     4
#define CDIMc  1152      // CONV_DIM
#define DPROJ  2208      // D_PROJ
#define MROWS  (Bc * Lc) // 32768

typedef __attribute__((ext_vector_type(16))) __bf16 v16bf;
typedef __attribute__((ext_vector_type(8)))  __bf16 v8bf;
typedef __attribute__((ext_vector_type(8)))  float  v8f;

__device__ __forceinline__ float siluf(float x) { return x / (1.0f + __expf(-x)); }
__device__ __forceinline__ float softplusf(float x) {
    if (x > 20.0f) return x;
    if (x < -20.0f) return __expf(x);
    return log1pf(__expf(x));
}

__device__ __forceinline__ v16bf load_op(const __bf16* p) {
    v8bf lo = *(const v8bf*)(p);
    v8bf hi = *(const v8bf*)(p + 16);
    v16bf v;
#pragma unroll
    for (int i = 0; i < 8; ++i) { v[i] = lo[i]; v[i + 8] = hi[i]; }
    return v;
}

// ---------------- element-wise / pack kernels ----------------

__global__ __launch_bounds__(256) void k_f32_to_bf16(const float* __restrict__ in,
                                                     __bf16* __restrict__ out, int n) {
    int i = blockIdx.x * 256 + threadIdx.x;
    if (i < n) out[i] = (__bf16)in[i];
}

// W (K,N) f32 row-major  ->  Wt (N,K) bf16 row-major
__global__ __launch_bounds__(256) void k_pack_wt(const float* __restrict__ W,
                                                 __bf16* __restrict__ Wt, int K, int N) {
    int i = blockIdx.x * 256 + threadIdx.x;
    if (i >= K * N) return;
    int k = i / N, n = i % N;
    Wt[(size_t)n * K + k] = (__bf16)W[i];
}

// txt[b,n] = text[b,0,:] @ W_txt + b_txt     (8 x 512, K=768 — tiny)
__global__ __launch_bounds__(256) void k_txt(const float* __restrict__ text,
                                             const float* __restrict__ W,
                                             const float* __restrict__ bias,
                                             float* __restrict__ txt) {
    int i = blockIdx.x * 256 + threadIdx.x;
    if (i >= Bc * DIMc) return;
    int b = i / DIMc, n = i % DIMc;
    const float* t = text + (size_t)b * LTc * TDIMc;   // [:,0,:]
    float s = bias[n];
    for (int k = 0; k < TDIMc; ++k) s += t[k] * W[k * DIMc + n];
    txt[i] = s;
}

// ---------------- bf16 WMMA GEMM (wave32, 32x32 register block / wave) ----------------
// C = A(MxK) * Bt(NxK)^T ; 2x2 tiles of 16x16 per wave -> 4 independent WMMAs / K-step.
// Epilogues:
//   EPI 0 : outF = acc + bias[n] + extra[(m/Lper)*N + n]        (txt broadcast)
//   EPI 1 : outF = acc + bias[n]
//   EPI 2 : outH = bf16(acc + bias[n] + extra[m*N + n])         (residual)
//   EPI 3 : column-sum over the 32-row block, atomicAdd accOut[(m/Lper)*N + n]
template <int EPI>
__global__ __launch_bounds__(256)
void k_wmma_gemm(const __bf16* __restrict__ A, const __bf16* __restrict__ Bt,
                 const float* __restrict__ bias, const float* __restrict__ extra,
                 float* __restrict__ outF, __bf16* __restrict__ outH,
                 float* __restrict__ accOut, int M, int N, int K, int Lper) {
    const int lane  = threadIdx.x & 31;
    const int wave  = blockIdx.x * 8 + (threadIdx.x >> 5);
    const int tilesN = N >> 5;                 // 32-wide N blocks
    const int tm = (wave / tilesN) << 5;
    const int tn = (wave % tilesN) << 5;
    if (tm >= M) return;

    const int half = lane >> 4;      // which K-half this lane carries
    const int r    = lane & 15;      // row of A / col of B / col of C

    const __bf16* aRow0 = A  + (size_t)(tm + r)      * K + half * 8;
    const __bf16* aRow1 = A  + (size_t)(tm + 16 + r) * K + half * 8;
    const __bf16* bCol0 = Bt + (size_t)(tn + r)      * K + half * 8;
    const __bf16* bCol1 = Bt + (size_t)(tn + 16 + r) * K + half * 8;

    v8f acc00 = {}, acc01 = {}, acc10 = {}, acc11 = {};
    for (int k0 = 0; k0 < K; k0 += 32) {
        v16bf a0 = load_op(aRow0 + k0);
        v16bf a1 = load_op(aRow1 + k0);
        v16bf b0 = load_op(bCol0 + k0);
        v16bf b1 = load_op(bCol1 + k0);
        acc00 = __builtin_amdgcn_wmma_f32_16x16x32_bf16(false, a0, false, b0, (short)0, acc00, false, false);
        acc01 = __builtin_amdgcn_wmma_f32_16x16x32_bf16(false, a0, false, b1, (short)0, acc01, false, false);
        acc10 = __builtin_amdgcn_wmma_f32_16x16x32_bf16(false, a1, false, b0, (short)0, acc10, false, false);
        acc11 = __builtin_amdgcn_wmma_f32_16x16x32_bf16(false, a1, false, b1, (short)0, acc11, false, false);
    }

    if constexpr (EPI == 3) {
        float s0 = 0.0f, s1 = 0.0f;
#pragma unroll
        for (int i = 0; i < 8; ++i) {
            s0 += acc00[i] + acc10[i];
            s1 += acc01[i] + acc11[i];
        }
        int b = tm / Lper;                      // 32-row blocks never straddle a batch
        atomicAdd(&accOut[(size_t)b * N + tn + r], s0);
        atomicAdd(&accOut[(size_t)b * N + tn + 16 + r], s1);
        return;
    } else {
        const v8f* accs[2][2] = {{&acc00, &acc01}, {&acc10, &acc11}};
#pragma unroll
        for (int mt = 0; mt < 2; ++mt) {
#pragma unroll
            for (int nt = 0; nt < 2; ++nt) {
                const v8f& a = *accs[mt][nt];
#pragma unroll
                for (int i = 0; i < 8; ++i) {
                    int m = tm + mt * 16 + half * 8 + i;
                    int n = tn + nt * 16 + r;
                    float v = a[i] + bias[n];
                    if constexpr (EPI == 0) v += extra[(size_t)(m / Lper) * N + n];
                    if constexpr (EPI == 2) v += extra[(size_t)m * N + n];
                    if constexpr (EPI == 2) outH[(size_t)m * N + n] = (__bf16)v;
                    else                    outF[(size_t)m * N + n] = v;
                }
            }
        }
    }
}

// ---------------- layernorm (one wave per 512-wide row) -> bf16 ----------------
__global__ __launch_bounds__(256)
void k_layernorm(const float* __restrict__ x, const float* __restrict__ w,
                 const float* __restrict__ b, __bf16* __restrict__ out, int rows) {
    int lane = threadIdx.x & 31;
    int row  = blockIdx.x * 8 + (threadIdx.x >> 5);
    if (row >= rows) return;
    const float* xr = x + (size_t)row * DIMc;
    float vals[16];
    float s = 0.0f;
#pragma unroll
    for (int i = 0; i < 16; ++i) { vals[i] = xr[lane + 32 * i]; s += vals[i]; }
#pragma unroll
    for (int off = 16; off > 0; off >>= 1) s += __shfl_xor(s, off, 32);
    float mu = s * (1.0f / DIMc);
    float vs = 0.0f;
#pragma unroll
    for (int i = 0; i < 16; ++i) { float d = vals[i] - mu; vs += d * d; }
#pragma unroll
    for (int off = 16; off > 0; off >>= 1) vs += __shfl_xor(vs, off, 32);
    float inv = rsqrtf(vs * (1.0f / DIMc) + 1e-5f);
#pragma unroll
    for (int i = 0; i < 16; ++i) {
        int c = lane + 32 * i;
        out[(size_t)row * DIMc + c] = (__bf16)((vals[i] - mu) * inv * w[c] + b[c]);
    }
}

// ---------------- causal depthwise conv (K=4) + SiLU + channel split ----------------
__global__ __launch_bounds__(256)
void k_conv_silu(const float* __restrict__ zx, const float* __restrict__ cw,
                 const float* __restrict__ cb, float* __restrict__ xh,
                 float* __restrict__ Bm, float* __restrict__ Cm) {
    long long i = (long long)blockIdx.x * 256 + threadIdx.x;
    if (i >= (long long)MROWS * CDIMc) return;
    int c = (int)(i % CDIMc);
    int m = (int)(i / CDIMc);
    int l = m & (Lc - 1);
    float s = cb[c];
#pragma unroll
    for (int k = 0; k < CK; ++k) {
        int ls = l + k - (CK - 1);
        if (ls >= 0)
            s += zx[(size_t)(m + k - (CK - 1)) * DPROJ + DIN + c] * cw[c * CK + k];
    }
    s = siluf(s);
    if (c < DIN)            xh[(size_t)m * DIN + c]        = s;
    else if (c < DIN + 64)  Bm[(size_t)m * 64 + (c - DIN)] = s;
    else                    Cm[(size_t)m * 64 + (c - DIN - 64)] = s;
}

// ---------------- dt = softplus(raw + bias); dA = exp(dt * -exp(A_log)) ----------------
__global__ __launch_bounds__(256)
void k_dt(const float* __restrict__ zx, const float* __restrict__ dt_bias,
          const float* __restrict__ A_log, float* __restrict__ dt, float* __restrict__ dA) {
    int i = blockIdx.x * 256 + threadIdx.x;
    if (i >= MROWS * NH) return;
    int h = i & (NH - 1);
    int m = i >> 5;
    float raw = zx[(size_t)m * DPROJ + (DIN + CDIMc) + h] + dt_bias[h];
    float d = softplusf(raw);
    float A = -__expf(A_log[h]);
    dt[i] = d;
    dA[i] = __expf(d * A);
}

// ---------------- Mamba2 selective scan: one block per (b, head) ----------------
// state h[32p x 64n] lives in registers (8 f32 / thread); x/B/C/dt/dA chunk-staged in LDS.
#define SCHUNK 16
__global__ __launch_bounds__(256)
void k_scan(const float* __restrict__ xh, const float* __restrict__ Bm,
            const float* __restrict__ Cm, const float* __restrict__ dt,
            const float* __restrict__ dA, const float* __restrict__ Dp,
            float* __restrict__ y) {
    const int b    = blockIdx.x >> 5;
    const int head = blockIdx.x & 31;
    const int t    = threadIdx.x;
    const int p    = t >> 3;          // head-dim row owned by this thread
    const int nb   = (t & 7) * 8;     // state columns nb..nb+7

    __shared__ float sx[SCHUNK * HD];
    __shared__ float sB[SCHUNK * DSTATE];
    __shared__ float sC[SCHUNK * DSTATE];
    __shared__ float sA[SCHUNK];
    __shared__ float sD[SCHUNK];

    float hreg[8];
#pragma unroll
    for (int j = 0; j < 8; ++j) hreg[j] = 0.0f;
    const float Dval = Dp[head];
    const size_t base = (size_t)b * Lc;

    for (int l0 = 0; l0 < Lc; l0 += SCHUNK) {
        // cooperative staging of SCHUNK steps of x (32), B (64), C (64), dA/dt
        for (int i = t; i < SCHUNK * HD; i += 256)
            sx[i] = xh[(base + l0 + (i >> 5)) * DIN + head * HD + (i & 31)];
        for (int i = t; i < SCHUNK * DSTATE; i += 256) {
            sB[i] = Bm[(base + l0 + (i >> 6)) * DSTATE + (i & 63)];
            sC[i] = Cm[(base + l0 + (i >> 6)) * DSTATE + (i & 63)];
        }
        if (t < SCHUNK)                      sA[t] = dA[(base + l0 + t) * NH + head];
        else if (t < 2 * SCHUNK)             sD[t - SCHUNK] = dt[(base + l0 + t - SCHUNK) * NH + head];
        __syncthreads();
#pragma unroll 1
        for (int s = 0; s < SCHUNK; ++s) {
            float a  = sA[s];
            float xp = sx[s * HD + p];
            float xb = sD[s] * xp;
            const float* Bs = sB + s * DSTATE + nb;
            const float* Cs = sC + s * DSTATE + nb;
            float part = 0.0f;
#pragma unroll
            for (int j = 0; j < 8; ++j) {
                hreg[j] = hreg[j] * a + xb * Bs[j];
                part += hreg[j] * Cs[j];
            }
            // reduce across the 8 lanes sharing p (lane bits 0..2)
#pragma unroll
            for (int off = 1; off < 8; off <<= 1) part += __shfl_xor(part, off, 32);
            if ((t & 7) == 0)
                y[(base + l0 + s) * DIN + head * HD + p] = part + Dval * xp;
        }
        __syncthreads();
    }
}

// ---------------- gating: yg = bf16( y * silu(z) ) ----------------
__global__ __launch_bounds__(256)
void k_gate(const float* __restrict__ y, const float* __restrict__ zx,
            __bf16* __restrict__ yg) {
    long long i = (long long)blockIdx.x * 256 + threadIdx.x;
    if (i >= (long long)MROWS * DIN) return;
    int c = (int)(i % DIN);
    int m = (int)(i / DIN);
    float z = zx[(size_t)m * DPROJ + c];
    yg[i] = (__bf16)(y[i] * siluf(z));
}

__global__ __launch_bounds__(256) void k_zero(float* __restrict__ p, int n) {
    int i = blockIdx.x * 256 + threadIdx.x;
    if (i < n) p[i] = 0.0f;
}

// out[b,n] = acc[b,n] / L + b_out[n]
__global__ __launch_bounds__(256)
void k_final(const float* __restrict__ acc, const float* __restrict__ b_out,
             float* __restrict__ out) {
    int i = blockIdx.x * 256 + threadIdx.x;
    if (i >= Bc * HIDc) return;
    out[i] = acc[i] * (1.0f / Lc) + b_out[i & (HIDc - 1)];
}

// ---------------- host launcher ----------------
extern "C" void kernel_launch(void* const* d_in, const int* in_sizes, int n_in,
                              void* d_out, int out_size, void* d_ws, size_t ws_size,
                              hipStream_t stream) {
    const float* image   = (const float*)d_in[0];
    const float* text    = (const float*)d_in[1];
    const float* W_in    = (const float*)d_in[2];
    const float* b_in    = (const float*)d_in[3];
    const float* W_txt   = (const float*)d_in[4];
    const float* b_txt   = (const float*)d_in[5];
    const float* norm_w  = (const float*)d_in[6];
    const float* norm_b  = (const float*)d_in[7];
    const float* W_zx    = (const float*)d_in[8];
    const float* b_zx    = (const float*)d_in[9];
    const float* conv_w  = (const float*)d_in[10];
    const float* conv_b  = (const float*)d_in[11];
    const float* dt_bias = (const float*)d_in[12];
    const float* A_log   = (const float*)d_in[13];
    const float* Dp      = (const float*)d_in[14];
    const float* W_blk   = (const float*)d_in[15];
    const float* b_blk   = (const float*)d_in[16];
    const float* W_out   = (const float*)d_in[17];
    const float* b_out   = (const float*)d_in[18];
    float* out = (float*)d_out;

    char* ws = (char*)d_ws;
    size_t off = 0;
    auto take = [&](size_t bytes) {
        size_t o = off;
        off = (off + bytes + 255) & ~(size_t)255;
        return o;
    };
    const size_t M = MROWS;
    float*  tok    = (float*) (ws + take(M * DIMc * 4));
    __bf16* xln    = (__bf16*)(ws + take(M * DIMc * 2));
    float*  zx     = (float*) (ws + take(M * (size_t)DPROJ * 4));
    float*  xh     = (float*) (ws + take(M * DIN * 4));
    float*  Bm     = (float*) (ws + take(M * 64 * 4));
    float*  Cm     = (float*) (ws + take(M * 64 * 4));
    float*  dt     = (float*) (ws + take(M * NH * 4));
    float*  dA     = (float*) (ws + take(M * NH * 4));
    float*  yb     = (float*) (ws + take(M * DIN * 4));
    __bf16* yg     = (__bf16*)(ws + take(M * DIN * 2));
    __bf16* tok2   = (__bf16*)(ws + take(M * DIMc * 2));
    float*  accb   = (float*) (ws + take(Bc * HIDc * 4));
    float*  txt    = (float*) (ws + take(Bc * DIMc * 4));
    __bf16* imgb   = (__bf16*)(ws + take(M * HIDc * 2));
    __bf16* wt_in  = (__bf16*)(ws + take((size_t)HIDc * DIMc * 2));  // (512,1024)
    __bf16* wt_zx  = (__bf16*)(ws + take((size_t)DIMc * DPROJ * 2)); // (2208,512)
    __bf16* wt_blk = (__bf16*)(ws + take((size_t)DIN * DIMc * 2));   // (512,1024)
    __bf16* wt_out = (__bf16*)(ws + take((size_t)DIMc * HIDc * 2));  // (1024,512)
    (void)ws_size; (void)in_sizes; (void)n_in; (void)out_size;

    auto blocks = [](long long n) { return (unsigned)((n + 255) / 256); };
    auto gemm_grid = [](long long Mm, long long Nn) {
        return (unsigned)(((Mm / 32) * (Nn / 32) + 7) / 8);
    };

    // weight packing (transpose to (N,K) bf16)
    k_pack_wt<<<blocks((long long)HIDc * DIMc), 256, 0, stream>>>(W_in,  wt_in,  HIDc, DIMc);
    k_pack_wt<<<blocks((long long)DIMc * DPROJ), 256, 0, stream>>>(W_zx,  wt_zx,  DIMc, DPROJ);
    k_pack_wt<<<blocks((long long)DIN  * DIMc), 256, 0, stream>>>(W_blk, wt_blk, DIN,  DIMc);
    k_pack_wt<<<blocks((long long)DIMc * HIDc), 256, 0, stream>>>(W_out, wt_out, DIMc, HIDc);
    // activation pack
    k_f32_to_bf16<<<blocks((long long)M * HIDc), 256, 0, stream>>>(image, imgb, (int)(M * HIDc));
    // text projection (tiny)
    k_txt<<<blocks(Bc * DIMc), 256, 0, stream>>>(text, W_txt, b_txt, txt);

    // GEMM1: tok = img @ W_in + b_in + txt[b]
    k_wmma_gemm<0><<<gemm_grid(M, DIMc), 256, 0, stream>>>(
        imgb, wt_in, b_in, txt, tok, nullptr, nullptr, (int)M, DIMc, HIDc, Lc);
    // layernorm -> bf16
    k_layernorm<<<(unsigned)((M + 7) / 8), 256, 0, stream>>>(tok, norm_w, norm_b, xln, (int)M);
    // GEMM2: zxbcdt = xln @ W_zxbcdt + b
    k_wmma_gemm<1><<<gemm_grid(M, DPROJ), 256, 0, stream>>>(
        xln, wt_zx, b_zx, nullptr, zx, nullptr, nullptr, (int)M, DPROJ, DIMc, Lc);
    // conv + silu + split
    k_conv_silu<<<blocks((long long)M * CDIMc), 256, 0, stream>>>(zx, conv_w, conv_b, xh, Bm, Cm);
    // dt / dA
    k_dt<<<blocks((long long)M * NH), 256, 0, stream>>>(zx, dt_bias, A_log, dt, dA);
    // selective scan: 256 blocks (b,head), 256 threads
    k_scan<<<Bc * NH, 256, 0, stream>>>(xh, Bm, Cm, dt, dA, Dp, yb);
    // gating -> bf16
    k_gate<<<blocks((long long)M * DIN), 256, 0, stream>>>(yb, zx, yg);
    // GEMM3: tok2 = bf16( tok + yg @ W_blk_out + b_blk )
    k_wmma_gemm<2><<<gemm_grid(M, DIMc), 256, 0, stream>>>(
        yg, wt_blk, b_blk, tok, nullptr, tok2, nullptr, (int)M, DIMc, DIN, Lc);
    // GEMM4 fused with mean: acc[b,:] += column sums of tok2 @ W_out
    k_zero<<<blocks(Bc * HIDc), 256, 0, stream>>>(accb, Bc * HIDc);
    k_wmma_gemm<3><<<gemm_grid(M, HIDc), 256, 0, stream>>>(
        tok2, wt_out, nullptr, nullptr, nullptr, nullptr, accb, (int)M, HIDc, DIMc, Lc);
    // out = acc / L + b_out
    k_final<<<blocks(Bc * HIDc), 256, 0, stream>>>(accb, b_out, out);
}
